// Encoder_46969762349095
// MI455X (gfx1250) — matrix operand order repaired and checked by reading, not compiled, not run on hardware
//
#include <hip/hip_runtime.h>
#include <hip/hip_bf16.h>

// ---------------------------------------------------------------------------
// Bidirectional LSTM encoder for MI455X (gfx1250, wave32, WMMA).
//   L=512, B=64, V=32000, E=512, H=512 (H2=256 per direction)
//
// Strategy:
//  1) input_gates_gemm: Gx[dir][l*B+b][0:1024] = emb[tok] @ W_ih^T + b
//     (big parallel bf16 WMMA GEMM, off the recurrence critical path)
//  2) lstm_recurrent: persistent kernel, 16 blocks (2 dirs x 8 groups of 32
//     hidden units). W_hh slice resident in LDS (bf16, 64KB). h exchanged as
//     bf16 via an L2-resident double-buffered global buffer, pulled into LDS
//     with GLOBAL_LOAD_ASYNC_TO_LDS_B128 (ASYNCcnt); grid barrier per step.
// ---------------------------------------------------------------------------

typedef __attribute__((ext_vector_type(16))) __bf16 v16bf;
typedef __attribute__((ext_vector_type(8)))  float  v8f;

#define L_SEQ 512
#define BATCH 64
#define EMB   512
#define H2D   256
#define G4    1024   // 4*H2

// float -> bf16, round-to-nearest-even (bit manipulation; no cvt dependency)
__device__ inline __bf16 f2bf(float f) {
    union { float f; unsigned u; } v; v.f = f;
    unsigned r = v.u + 0x7FFFu + ((v.u >> 16) & 1u);
    unsigned short hs = (unsigned short)(r >> 16);
    __bf16 out; __builtin_memcpy(&out, &hs, sizeof(out));
    return out;
}

// A-matrix fragment (16x32 bf16, MxK) per ISA 7.12.2:
// lanes 0-15: M=lane, K pairs {0..7} then {16..23}; lanes 16-31: K {8..15},{24..31}
__device__ inline v16bf frag_a(const __bf16* A, int lda, int row0, int lane) {
    int m  = row0 + (lane & 15);
    int kh = (lane >> 4) << 3;        // 0 or 8
    v16bf a;
#pragma unroll
    for (int i = 0; i < 8; ++i) a[i]     = A[m * lda + kh + i];
#pragma unroll
    for (int i = 0; i < 8; ++i) a[8 + i] = A[m * lda + 16 + kh + i];
    return a;
}

// B-matrix fragment (32x16 bf16, KxN): lane = N column (mod 16),
// lanes 0-15 hold K {0..7},{16..23}; lanes 16-31 hold K {8..15},{24..31}.
// Bm is stored row-major as W[n][k] (i.e. B transposed), ldb = K stride.
__device__ inline v16bf frag_b(const __bf16* Bm, int ldb, int col0, int lane) {
    int n  = col0 + (lane & 15);
    int k0 = (lane >> 4) << 3;        // 0 or 8
    v16bf b;
#pragma unroll
    for (int i = 0; i < 8; ++i) b[i]     = Bm[n * ldb + k0 + i];
#pragma unroll
    for (int i = 0; i < 8; ++i) b[8 + i] = Bm[n * ldb + 16 + k0 + i];
    return b;
}

// ---------------------------------------------------------------------------
// Kernel 1: gate pre-activations from the embedding gather (both directions).
// Grid: (32768/64, 1024/64, 2). Block: 256 threads = 8 waves.
// Each block computes a 64x64 tile; each wave two 16x16 WMMA tiles.
// ---------------------------------------------------------------------------
__global__ void __launch_bounds__(256)
input_gates_gemm(const int*   __restrict__ tokens,
                 const float* __restrict__ emb,
                 const float* __restrict__ w_ih_f, const float* __restrict__ b_f,
                 const float* __restrict__ w_ih_b, const float* __restrict__ b_b,
                 float*       __restrict__ Gx)
{
    __shared__ __bf16 As[64 * 32];
    __shared__ __bf16 Bs[64 * 32];

    const int    dir  = blockIdx.z;
    const float* W    = dir ? w_ih_b : w_ih_f;
    const float* bias = dir ? b_b    : b_f;
    float*       G    = Gx + (size_t)dir * (size_t)(L_SEQ * BATCH) * G4;

    const int m0   = blockIdx.x * 64;   // flattened (l*B+b) row tile
    const int n0   = blockIdx.y * 64;   // gate column tile
    const int tid  = threadIdx.x;
    const int lane = tid & 31;
    const int wave = tid >> 5;

    const int t0  = wave * 2, t1 = wave * 2 + 1;     // 16 tiles / 8 waves
    const int mt0 = t0 >> 2, nt0 = t0 & 3;
    const int mt1 = t1 >> 2, nt1 = t1 & 3;
    v8f acc0 = {}; v8f acc1 = {};

    for (int k0 = 0; k0 < EMB; k0 += 32) {
        __syncthreads();
        // Stage A (gathered embedding rows), 64x32 -> bf16 LDS
#pragma unroll
        for (int e = 0; e < 8; ++e) {
            int idx = tid + e * 256;               // 0..2047
            int r = idx >> 5, c = idx & 31;
            int tok = tokens[m0 + r];
            As[r * 32 + c] = f2bf(emb[(size_t)tok * EMB + k0 + c]);
        }
        // Stage B (W_ih rows n0..n0+63, K chunk) -> bf16 LDS
#pragma unroll
        for (int e = 0; e < 8; ++e) {
            int idx = tid + e * 256;
            int r = idx >> 5, c = idx & 31;
            Bs[r * 32 + c] = f2bf(W[(size_t)(n0 + r) * EMB + k0 + c]);
        }
        __syncthreads();

        v16bf a0 = frag_a(As, 32, mt0 * 16, lane);
        v16bf b0 = frag_b(Bs, 32, nt0 * 16, lane);
        acc0 = __builtin_amdgcn_wmma_f32_16x16x32_bf16(false, a0, false, b0,
                                                       (short)0, acc0, false, false);
        v16bf a1 = frag_a(As, 32, mt1 * 16, lane);
        v16bf b1 = frag_b(Bs, 32, nt1 * 16, lane);
        acc1 = __builtin_amdgcn_wmma_f32_16x16x32_bf16(false, a1, false, b1,
                                                       (short)0, acc1, false, false);
    }

    // C layout: VGPR r, lane -> M = r + 8*(lane>=16), N = lane&15. Fold bias here.
    {
        int n  = n0 + nt0 * 16 + (lane & 15);
        int mb = m0 + mt0 * 16 + ((lane >> 4) << 3);
        float bv = bias[n];
#pragma unroll
        for (int r = 0; r < 8; ++r)
            G[(size_t)(mb + r) * G4 + n] = acc0[r] + bv;
    }
    {
        int n  = n0 + nt1 * 16 + (lane & 15);
        int mb = m0 + mt1 * 16 + ((lane >> 4) << 3);
        float bv = bias[n];
#pragma unroll
        for (int r = 0; r < 8; ++r)
            G[(size_t)(mb + r) * G4 + n] = acc1[r] + bv;
    }
}

// ---------------------------------------------------------------------------
// Grid barrier (persistent kernel, deterministic: counters re-zeroed per launch)
// ---------------------------------------------------------------------------
__device__ inline void grid_sync(unsigned* syncv, unsigned nb, unsigned* phase)
{
    __syncthreads();
    if (threadIdx.x == 0) {
        unsigned my = *phase + 1;
        __threadfence();
        unsigned prev = atomicAdd(&syncv[0], 1u);
        if (prev == nb - 1) {
            syncv[0] = 0;
            __threadfence();
            atomicExch(&syncv[1], my);
        } else {
            while (atomicAdd(&syncv[1], 0u) < my) {
                __builtin_amdgcn_s_sleep(1);   // back off the spin
            }
        }
        *phase = my;
    }
    __syncthreads();
}

// ---------------------------------------------------------------------------
// Kernel 2: recurrence. 16 persistent blocks = 2 dirs x 8 hidden groups of 32.
// Per step: gh(64x128) = h(64x256) @ Wslice^T(128x256) via bf16 WMMA,
// then fused gate nonlinearities / cell update / masked h write.
// h state lives in global memory as bf16 (L2-resident, 32KB/dir) and is pulled
// straight into LDS with GLOBAL_LOAD_ASYNC_TO_LDS_B128 (no VGPR round trip).
// ---------------------------------------------------------------------------
__global__ void __launch_bounds__(256)
lstm_recurrent(const float*  __restrict__ mask,
               const float*  __restrict__ w_hh_f,
               const float*  __restrict__ w_hh_b,
               const float*  __restrict__ Gx,
               __bf16*       __restrict__ hbuf,   // [2 phase][2 dir][64][256] bf16
               unsigned*     __restrict__ syncv,  // [0]=cnt, [1]=gen
               float*        __restrict__ out)    // [L][B][512]
{
    extern __shared__ char smem[];
    __bf16* Ws  = (__bf16*)smem;                               // [128][256]  64KB
    __bf16* hbf = (__bf16*)(smem + 128 * 256 * 2);             // [64][256]   32KB
    float*  gh  = (float*)(smem + 128 * 256 * 2 + 64 * 256 * 2); // [64][128] 32KB
    float*  cst = (float*)((char*)gh + 64 * 128 * 4);          // [64][32]     8KB

    const int dir  = blockIdx.x >> 3;
    const int grp  = blockIdx.x & 7;
    const int base = grp * 32;                 // hidden units [base, base+32)
    const int tid  = threadIdx.x;
    const int lane = tid & 31;
    const int wave = tid >> 5;

    const float* Whh = dir ? w_hh_b : w_hh_f;
    const float* Gd  = Gx + (size_t)dir * (size_t)(L_SEQ * BATCH) * G4;

    // Resident W_hh slice: slice row (gate*32+j) <- W_hh row (gate*256+base+j)
    for (int e = tid; e < 128 * 256; e += 256) {
        int r = e >> 8, k = e & 255;
        int gate = r >> 5, j = r & 31;
        Ws[e] = f2bf(Whh[(size_t)(gate * 256 + base + j) * H2D + k]);
    }
    for (int e = tid; e < 64 * 32; e += 256) cst[e] = 0.f;   // c state
    __syncthreads();

    unsigned phase = 0;
    const unsigned NB = gridDim.x;

    for (int t = 0; t < L_SEQ; ++t) {
        grid_sync(syncv, NB, &phase);          // prior step's h fully visible

        const int l = dir ? (L_SEQ - 1 - t) : t;
        const __bf16* hin  = hbuf + ((size_t)((t)     & 1) * 2 + dir) * (BATCH * H2D);
        __bf16*       hout = hbuf + ((size_t)((t + 1) & 1) * 2 + dir) * (BATCH * H2D);

        // Prefetch this step's gate pre-activations (consumed after the GEMM):
        // one lane per wave issues global_prefetch for the 4 gate segments.
        if ((tid & 31) == 0) {
            for (int e = tid; e < BATCH * 32; e += 256) {
                const float* g = Gd + ((size_t)l * BATCH + (e >> 5)) * G4 + base;
                __builtin_prefetch(g,       0, 0);
                __builtin_prefetch(g + 256, 0, 0);
                __builtin_prefetch(g + 512, 0, 0);
                __builtin_prefetch(g + 768, 0, 0);
            }
        }

        // h_prev (bf16, 32KB) -> LDS via async copy: L2 -> LDS, no VGPRs.
        // 256 threads x 16B x 8 rounds = 32768B. Tracked with ASYNCcnt.
#pragma unroll
        for (int r = 0; r < 8; ++r) {
            unsigned lofs = (unsigned)(size_t)(const void*)
                            ((const char*)hbf + r * 4096 + tid * 16);
            unsigned long long gaddr = (unsigned long long)(size_t)
                            ((const char*)hin + r * 4096 + tid * 16);
            asm volatile("global_load_async_to_lds_b128 %0, %1, off"
                         :: "v"(lofs), "v"(gaddr) : "memory");
        }
        asm volatile("s_wait_asynccnt 0" ::: "memory");
        __syncthreads();

        // WMMA: wave w owns N-tile w (16 cols), all 4 M tiles; K = 256 in 8 steps
        v8f acc[4] = {};
#pragma unroll
        for (int kk = 0; kk < H2D; kk += 32) {
            v16bf bfrag = frag_b(Ws + kk, 256, wave * 16, lane);
#pragma unroll
            for (int mt = 0; mt < 4; ++mt) {
                v16bf afrag = frag_a(hbf + kk, 256, mt * 16, lane);
                acc[mt] = __builtin_amdgcn_wmma_f32_16x16x32_bf16(
                    false, afrag, false, bfrag, (short)0, acc[mt], false, false);
            }
        }
#pragma unroll
        for (int mt = 0; mt < 4; ++mt) {
            int n  = wave * 16 + (lane & 15);
            int mb = mt * 16 + ((lane >> 4) << 3);
#pragma unroll
            for (int r = 0; r < 8; ++r) gh[(mb + r) * 128 + n] = acc[mt][r];
        }
        __syncthreads();

        // Fused gates + cell update + masked h for this block's 32 hidden units
        for (int e = tid; e < BATCH * 32; e += 256) {
            int b = e >> 5, j = e & 31;
            const float* g = Gd + ((size_t)l * BATCH + b) * G4;
            float iv = gh[b * 128 +  0 + j] + g[0 * 256 + base + j];
            float fv = gh[b * 128 + 32 + j] + g[1 * 256 + base + j];
            float gv = gh[b * 128 + 64 + j] + g[2 * 256 + base + j];
            float ov = gh[b * 128 + 96 + j] + g[3 * 256 + base + j];
            float si = 1.f / (1.f + __expf(-iv));
            float sf = 1.f / (1.f + __expf(-fv));
            float so = 1.f / (1.f + __expf(-ov));
            float c  = sf * cst[e] + si * tanhf(gv);
            float h  = so * tanhf(c);
            float m  = mask[l * BATCH + b];
            h *= m; c *= m;
            cst[e] = c;
            hout[b * H2D + base + j] = f2bf(h);   // bf16 state exchange
            out[((size_t)l * BATCH + b) * 512 + dir * H2D + base + j] = h;
        }
        __threadfence();
        __syncthreads();
    }
}

// ---------------------------------------------------------------------------
// Workspace init (graph-replay safe: runs every launch, same stream)
// ---------------------------------------------------------------------------
__global__ void init_ws(unsigned* syncv, unsigned short* hbuf)
{
    int tid = blockIdx.x * blockDim.x + threadIdx.x;
    if (tid < 2) syncv[tid] = 0u;
    for (int i = tid; i < 4 * BATCH * H2D; i += blockDim.x * gridDim.x)
        hbuf[i] = 0;                            // bf16 +0.0
}

// ---------------------------------------------------------------------------
extern "C" void kernel_launch(void* const* d_in, const int* in_sizes, int n_in,
                              void* d_out, int out_size, void* d_ws, size_t ws_size,
                              hipStream_t stream)
{
    const int*   tokens = (const int*)  d_in[0];
    const float* mask   = (const float*)d_in[1];
    const float* emb    = (const float*)d_in[2];
    const float* w_ih_f = (const float*)d_in[3];
    const float* w_hh_f = (const float*)d_in[4];
    const float* b_f    = (const float*)d_in[5];
    const float* w_ih_b = (const float*)d_in[6];
    const float* w_hh_b = (const float*)d_in[7];
    const float* b_b    = (const float*)d_in[8];
    float*       out    = (float*)d_out;

    // Workspace layout
    char* ws = (char*)d_ws;
    unsigned* syncv = (unsigned*)ws;                     // 256 B
    __bf16*   hbuf  = (__bf16*)(ws + 256);               // 2*2*64*256 bf16 = 128 KB
    float*    Gx    = (float*)(ws + 256 + 4 * BATCH * H2D * sizeof(unsigned short));

    init_ws<<<64, 256, 0, stream>>>(syncv, (unsigned short*)hbuf);

    dim3 gA((L_SEQ * BATCH) / 64, G4 / 64, 2);
    input_gates_gemm<<<gA, 256, 0, stream>>>(tokens, emb, w_ih_f, b_f,
                                             w_ih_b, b_b, Gx);

    size_t lds = (size_t)128 * 256 * 2   // Ws
               + (size_t)64  * 256 * 2   // hbf
               + (size_t)64  * 128 * 4   // gh
               + (size_t)64  * 32  * 4;  // cst   => 136 KB (within 320 KB/WGP)
    lstm_recurrent<<<16, 256, lds, stream>>>(mask, w_hh_f, w_hh_b, Gx,
                                             hbuf, syncv, out);
}
